// DynamicRouter_24807731101934
// MI455X (gfx1250) — matrix operand order repaired and checked by reading, not compiled, not run on hardware
//
#include <hip/hip_runtime.h>
#include <math.h>

#define NTOK 16384
#define DDIM 2048
#define HDIM 4096
#define EDIM 64

typedef __attribute__((ext_vector_type(16))) _Float16 v16h;
typedef __attribute__((ext_vector_type(8)))  float    v8f;

union FragH {
    int4   q[2];
    v16h   v;
};

union Pack4H {
    _Float16 h[4];
    uint2    u;
};

// ---------------------------------------------------------------------------
// Kernel 0: W2 [H,E] f32 -> W2T [E,H] f16 (K-contiguous for WMMA B fragments)
// ---------------------------------------------------------------------------
__global__ void k0_w2_transpose(const float* __restrict__ W2,
                                _Float16* __restrict__ W2T) {
    int idx = blockIdx.x * blockDim.x + threadIdx.x;   // over H*E
    int k = idx / EDIM;
    int e = idx % EDIM;
    W2T[(size_t)e * HDIM + k] = (_Float16)W2[idx];
}

// ---------------------------------------------------------------------------
// Kernel 1: h = gelu(x @ W1 + b1), stored f16.
// Block = 256 threads = 8 waves; block tile 128x128; wave tile 64x32 (4x2 WMMA).
// LDS: 32 KB buffer; [0,8K) = A tile [row][k], [8K,16K) = B tile [col][k]
// during the K loop; whole 32 KB = 128x128 f16 output tile in the epilogue.
// ---------------------------------------------------------------------------
__global__ __launch_bounds__(256, 2)
void k1_gemm1_gelu(const float* __restrict__ x,
                   const float* __restrict__ W1,
                   const float* __restrict__ b1,
                   _Float16* __restrict__ hbf) {
    __shared__ __align__(16) _Float16 smem[128 * 128];   // 32 KB
    _Float16* lsA = smem;               // [row][k], 128x32
    _Float16* lsB = smem + 128 * 32;    // [col][k], 128x32

    const int tid   = threadIdx.x;
    const int lane  = tid & 31;
    const int wave  = tid >> 5;
    const int l16   = lane & 15;
    const int khalf = lane >> 4;

    const int rowBase = blockIdx.y * 128;
    const int colBase = blockIdx.x * 128;
    const int wm = (wave & 1) * 64;    // wave row offset
    const int wn = (wave >> 1) * 32;   // wave col offset

    v8f acc[4][2];
    v8f zero = {};
    #pragma unroll
    for (int mi = 0; mi < 4; ++mi)
        #pragma unroll
        for (int ni = 0; ni < 2; ++ni) acc[mi][ni] = zero;

    for (int k0 = 0; k0 < DDIM; k0 += 32) {
        // ---- stage A tile: x[rowBase..+128, k0..+32] f32 -> f16 row-major.
        // Each thread: one float4 (4 K of one row) -> packed ds_store_b64.
        #pragma unroll
        for (int i = 0; i < 4; ++i) {
            int f = tid + i * 256;            // float4 id, 0..1023
            int r = f >> 3;                   // 0..127
            int c = (f & 7) << 2;             // 0..28
            const float4 v = *(const float4*)(x + (size_t)(rowBase + r) * DDIM + k0 + c);
            Pack4H p;
            p.h[0] = (_Float16)v.x; p.h[1] = (_Float16)v.y;
            p.h[2] = (_Float16)v.z; p.h[3] = (_Float16)v.w;
            *(uint2*)&lsA[r * 32 + c] = p.u;
        }
        // ---- stage B tile transposed: W1[k0..+32, colBase..+128] -> lsB[col][k].
        // Each thread: 4 K-consecutive values of ONE column (4 coalesced b32
        // loads, stride H) -> packed ds_store_b64 (K-contiguous in LDS).
        #pragma unroll
        for (int i = 0; i < 4; ++i) {
            int f  = tid + i * 256;           // 0..1023
            int c  = f & 127;                 // column 0..127
            int kq = f >> 7;                  // k-quad 0..7
            const float* wp = W1 + (size_t)(k0 + kq * 4) * HDIM + colBase + c;
            Pack4H p;
            p.h[0] = (_Float16)wp[0];
            p.h[1] = (_Float16)wp[HDIM];
            p.h[2] = (_Float16)wp[2 * HDIM];
            p.h[3] = (_Float16)wp[3 * HDIM];
            *(uint2*)&lsB[c * 32 + kq * 4] = p.u;
        }
        __syncthreads();

        // ---- fragments per ISA 16-bit layouts
        FragH afr[4], bfr[2];
        #pragma unroll
        for (int mi = 0; mi < 4; ++mi) {
            int r = wm + mi * 16 + l16;       // A: row = lane%16, K halves by lane/16
            afr[mi].q[0] = *(const int4*)&lsA[r * 32 + khalf * 8];        // K base..base+7
            afr[mi].q[1] = *(const int4*)&lsA[r * 32 + khalf * 8 + 16];   // K base+16..+23
        }
        #pragma unroll
        for (int ni = 0; ni < 2; ++ni) {
            int c = wn + ni * 16 + l16;       // B: col = lane%16, K halves by lane/16 (same as A)
            bfr[ni].q[0] = *(const int4*)&lsB[c * 32 + khalf * 8];
            bfr[ni].q[1] = *(const int4*)&lsB[c * 32 + khalf * 8 + 16];
        }
        #pragma unroll
        for (int mi = 0; mi < 4; ++mi)
            #pragma unroll
            for (int ni = 0; ni < 2; ++ni)
                acc[mi][ni] = __builtin_amdgcn_wmma_f32_16x16x32_f16(
                    false, afr[mi].v, false, bfr[ni].v,
                    (short)0, acc[mi][ni], false, false);
        __syncthreads();
    }

    // ---- epilogue: +b1, exact GELU, stage f16 tile in LDS, coalesced b128 out
    #pragma unroll
    for (int ni = 0; ni < 2; ++ni) {
        int colL = wn + ni * 16 + l16;                  // local col 0..127
        float bias = b1[colBase + colL];
        #pragma unroll
        for (int mi = 0; mi < 4; ++mi) {
            int rL = wm + mi * 16 + khalf * 8;          // D: VGPR r -> M=r (+8 hi half)
            #pragma unroll
            for (int r = 0; r < 8; ++r) {
                float v = acc[mi][ni][r] + bias;
                float g = 0.5f * v * (1.0f + erff(v * 0.70710678118654752f));
                smem[(rL + r) * 128 + colL] = (_Float16)g;
            }
        }
    }
    __syncthreads();
    #pragma unroll
    for (int i = 0; i < 8; ++i) {
        int f   = tid + i * 256;          // int4 id, 0..2047
        int row = f >> 4;                 // 0..127
        int c8  = (f & 15) << 3;          // elem 0..120
        *(int4*)(hbf + (size_t)(rowBase + row) * HDIM + colBase + c8) =
            *(const int4*)&smem[row * 128 + c8];
    }
}

// ---------------------------------------------------------------------------
// Kernel 2: logits = h @ W2 + b2 -> softmax -> slices + top-2.
// One wave per 16 rows; 4 WMMA tiles cover all 64 experts; A/B fragments are
// direct 16B-aligned global loads (W2T is L2-resident).
// ---------------------------------------------------------------------------
__global__ __launch_bounds__(256, 4)
void k2_gemm2_softmax_topk(const _Float16* __restrict__ hbf,
                           const _Float16* __restrict__ W2T,
                           const float* __restrict__ b2,
                           float* __restrict__ outG,
                           float* __restrict__ outLW,
                           float* __restrict__ outLI,
                           float* __restrict__ outW) {
    const int tid   = threadIdx.x;
    const int lane  = tid & 31;
    const int wave  = tid >> 5;
    const int l16   = lane & 15;
    const int khalf = lane >> 4;
    const int rowBase = blockIdx.x * 128 + wave * 16;

    v8f acc[4];
    v8f zero = {};
    #pragma unroll
    for (int t = 0; t < 4; ++t) acc[t] = zero;

    const int arow = rowBase + l16;
    for (int k0 = 0; k0 < HDIM; k0 += 32) {
        FragH a;
        const _Float16* ap = hbf + (size_t)arow * HDIM + k0 + khalf * 8;
        a.q[0] = *(const int4*)ap;
        a.q[1] = *(const int4*)(ap + 16);
        #pragma unroll
        for (int t = 0; t < 4; ++t) {
            FragH b;
            const _Float16* bp = W2T + (size_t)(t * 16 + l16) * HDIM + k0 + khalf * 8;
            b.q[0] = *(const int4*)bp;
            b.q[1] = *(const int4*)(bp + 16);
            acc[t] = __builtin_amdgcn_wmma_f32_16x16x32_f16(
                false, a.v, false, b.v, (short)0, acc[t], false, false);
        }
    }

    float b2v[4];
    #pragma unroll
    for (int t = 0; t < 4; ++t) b2v[t] = b2[t * 16 + l16];

    // Each lane holds 4 logits per row (cols t*16 + l16); row r lives in
    // lanes 0-15 (rows rbase+0..7) and lanes 16-31 (rows rbase+8..15).
    #pragma unroll
    for (int r = 0; r < 8; ++r) {
        float wv[4];
        #pragma unroll
        for (int t = 0; t < 4; ++t) wv[t] = acc[t][r] + b2v[t];

        // softmax (max-subtracted); width-16 xor reductions stay within halves
        float mx = fmaxf(fmaxf(wv[0], wv[1]), fmaxf(wv[2], wv[3]));
        #pragma unroll
        for (int off = 1; off < 16; off <<= 1)
            mx = fmaxf(mx, __shfl_xor(mx, off, 32));
        float ex[4], s = 0.0f;
        #pragma unroll
        for (int t = 0; t < 4; ++t) { ex[t] = __expf(wv[t] - mx); s += ex[t]; }
        #pragma unroll
        for (int off = 1; off < 16; off <<= 1)
            s += __shfl_xor(s, off, 32);
        float inv = 1.0f / s;
        #pragma unroll
        for (int t = 0; t < 4; ++t) wv[t] = ex[t] * inv;

        const int row = rowBase + r + khalf * 8;

        // full weights [N,64]
        #pragma unroll
        for (int t = 0; t < 4; ++t)
            outW[(size_t)row * 64 + t * 16 + l16] = wv[t];
        // global (shared) weights: cols 0..7
        if (l16 < 8) outG[(size_t)row * 8 + l16] = wv[0];

        // top-2 over local cols 8..63 (local index = col - 8); tie -> lower idx
        float bv = -INFINITY; int bi = 1 << 20;
        #pragma unroll
        for (int t = 0; t < 4; ++t) {
            int col = t * 16 + l16;
            if (col >= 8) {
                int idx = col - 8;
                if (wv[t] > bv || (wv[t] == bv && idx < bi)) { bv = wv[t]; bi = idx; }
            }
        }
        #pragma unroll
        for (int off = 1; off < 16; off <<= 1) {
            float ov = __shfl_xor(bv, off, 32);
            int   oi = __shfl_xor(bi, off, 32);
            if (ov > bv || (ov == bv && oi < bi)) { bv = ov; bi = oi; }
        }
        float sv = -INFINITY; int si = 1 << 20;
        #pragma unroll
        for (int t = 0; t < 4; ++t) {
            int col = t * 16 + l16;
            if (col >= 8) {
                int idx = col - 8;
                if (idx != bi && (wv[t] > sv || (wv[t] == sv && idx < si))) { sv = wv[t]; si = idx; }
            }
        }
        #pragma unroll
        for (int off = 1; off < 16; off <<= 1) {
            float ov = __shfl_xor(sv, off, 32);
            int   oi = __shfl_xor(si, off, 32);
            if ((oi != bi) && (ov > sv || (ov == sv && oi < si))) { sv = ov; si = oi; }
        }
        if (l16 == 0) {
            outLW[(size_t)row * 2 + 0] = bv;
            outLW[(size_t)row * 2 + 1] = sv;
            outLI[(size_t)row * 2 + 0] = (float)bi;
            outLI[(size_t)row * 2 + 1] = (float)si;
        }
    }
}

// ---------------------------------------------------------------------------
extern "C" void kernel_launch(void* const* d_in, const int* in_sizes, int n_in,
                              void* d_out, int out_size, void* d_ws, size_t ws_size,
                              hipStream_t stream) {
    (void)in_sizes; (void)n_in; (void)out_size; (void)ws_size;

    const float* x  = (const float*)d_in[0];
    const float* W1 = (const float*)d_in[1];
    const float* b1 = (const float*)d_in[2];
    const float* W2 = (const float*)d_in[3];
    const float* b2 = (const float*)d_in[4];

    float* out   = (float*)d_out;
    float* outG  = out;                          // [N, 8]
    float* outLW = out + (size_t)NTOK * 8;       // [N, 2]
    float* outLI = out + (size_t)NTOK * 10;      // [N, 2] (indices as float)
    float* outW  = out + (size_t)NTOK * 12;      // [N, 64]

    _Float16* hbf = (_Float16*)d_ws;                         // N*H f16 = 128 MB
    _Float16* w2t = hbf + (size_t)NTOK * HDIM;               // E*H f16 = 512 KB

    k0_w2_transpose<<<(HDIM * EDIM) / 256, 256, 0, stream>>>(W2, w2t);

    dim3 g1(HDIM / 128, NTOK / 128);
    k1_gemm1_gelu<<<g1, 256, 0, stream>>>(x, W1, b1, hbf);

    k2_gemm2_softmax_topk<<<NTOK / 128, 256, 0, stream>>>(hbf, w2t, b2,
                                                          outG, outLW, outLI, outW);
}
